// IMGMMadapter_45861660786838
// MI455X (gfx1250) — compile-verified, matrix operand order checked by reading, not run on hardware
//
#include <hip/hip_runtime.h>
#include <hip/hip_bf16.h>

typedef __attribute__((ext_vector_type(16))) _Float16 v16h;
typedef __attribute__((ext_vector_type(4)))  _Float16 v4h;
typedef __attribute__((ext_vector_type(8)))  float    v8f;

#define WMMA_F16(a, b, c) \
  __builtin_amdgcn_wmma_f32_16x16x32_f16(false, (a), false, (b), (short)0, (c), false, false)

// ---------------------------------------------------------------------------
// Fragment loaders for V_WMMA_F32_16X16X32_F16 (wave32).
// 16-bit A (16xK slice) / B^T (N rows x K): lane covers row/col (lane&15),
// 16 halves at K offsets {kb..kb+7, kb+16..kb+23}, kb = 8*(lane>=16).
// ---------------------------------------------------------------------------
__device__ __forceinline__ v16h frag_ld_k(const _Float16* rowbase, int lane) {
  const _Float16* p = rowbase + ((lane & 16) >> 1);
  v16h v;
#pragma unroll
  for (int e = 0; e < 8; ++e) v[e] = p[e];
#pragma unroll
  for (int e = 0; e < 8; ++e) v[e + 8] = p[16 + e];
  return v;
}

__device__ __forceinline__ v16h frag_ld(const _Float16* base, int stride, int lane) {
  return frag_ld_k(base + (lane & 15) * stride, lane);
}

// scan-index -> hw-position maps for the 4 SS2D directions
__device__ __forceinline__ int pos_dir(int k, int l) {
  switch (k) {
    case 0:  return l;
    case 1:  return (l % 7) * 7 + l / 7;
    case 2:  return 48 - l;
    default: { int lr = 48 - l; return (lr % 7) * 7 + lr / 7; }
  }
}

__device__ __forceinline__ v4h cvt4(float a, float b, float c, float d) {
  v4h h;
  h[0] = (_Float16)a; h[1] = (_Float16)b; h[2] = (_Float16)c; h[3] = (_Float16)d;
  return h;
}

// ---------------------------------------------------------------------------
// Kernel 0: convert weight matrices fp32 -> f16 (all stored N x K row-major)
// ---------------------------------------------------------------------------
__global__ void k_wconv(const float* __restrict__ Wd, const float* __restrict__ Wu,
                        const float* __restrict__ inw, const float* __restrict__ outw,
                        _Float16* __restrict__ wd_h, _Float16* __restrict__ wu_h,
                        _Float16* __restrict__ inw_h, _Float16* __restrict__ outw_h) {
  int i = blockIdx.x * blockDim.x + threadIdx.x;
  if (i < 128 * 1024) wd_h[i]  = (_Float16)Wd[i];
  if (i < 1024 * 128) wu_h[i]  = (_Float16)Wu[i];
  if (i < 512 * 128)  inw_h[i] = (_Float16)inw[i];
  if (i < 128 * 256)  outw_h[i] = (_Float16)outw[i];
}

// ---------------------------------------------------------------------------
// Kernel 1: xmid[b*49+s][n] = GELU( x[s][b][:] . Wd[n][:] + bd[n] )
// M = 50176 (64/block), N = 128, K = 1024; double-buffered 32-wide A chunks
// ---------------------------------------------------------------------------
__global__ __launch_bounds__(256) void k_downproj(const float* __restrict__ x,
                                                  const _Float16* __restrict__ wd_h,
                                                  const float* __restrict__ bd,
                                                  float* __restrict__ xmid) {
  __shared__ _Float16 As[2][64 * 32];
  const int tid = threadIdx.x;
  const int lane = tid & 31;
  const int wv = tid >> 5;
  const int mq = wv & 3, nq = wv >> 2;         // 4 M-subtiles x 2 N-subtiles(64)
  const int mbase = blockIdx.x * 64;

  // per-thread streaming source (row gather from (S,B,HID)): row r, 8-float slot q
  const int r = tid >> 2;
  const int q = (tid & 3) * 8;
  {
    int gr = mbase + r;
    int bb = gr / 49, s = gr - bb * 49;
    const float* srcbase = x + ((size_t)s * 1024 + bb) * 1024 + q;

    // preload chunk 0
    {
      const float4* s4 = (const float4*)srcbase;
      float4 u = s4[0], w = s4[1];
      *(v4h*)&As[0][r * 32 + q]     = cvt4(u.x, u.y, u.z, u.w);
      *(v4h*)&As[0][r * 32 + q + 4] = cvt4(w.x, w.y, w.z, w.w);
    }

    v8f acc[4] = {};
    for (int c = 0; c < 32; ++c) {
      __syncthreads();
      if (c + 1 < 32) {
        const float* src = srcbase + (c + 1) * 32;
        __builtin_prefetch(src + 64, 0, 1);           // 2 chunks ahead
        const float4* s4 = (const float4*)src;
        float4 u = s4[0], w = s4[1];
        _Float16* dst = &As[(c + 1) & 1][r * 32 + q];
        *(v4h*)dst       = cvt4(u.x, u.y, u.z, u.w);
        *(v4h*)(dst + 4) = cvt4(w.x, w.y, w.z, w.w);
      }
      v16h a = frag_ld(As[c & 1] + mq * 16 * 32, 32, lane);
      v16h bf[4];
#pragma unroll
      for (int i = 0; i < 4; ++i)
        bf[i] = frag_ld_k(wd_h + (size_t)(nq * 64 + i * 16 + (lane & 15)) * 1024 + c * 32, lane);
#pragma unroll
      for (int i = 0; i < 4; ++i) acc[i] = WMMA_F16(a, bf[i], acc[i]);
    }

#pragma unroll
    for (int i = 0; i < 4; ++i) {
      int n = nq * 64 + i * 16 + (lane & 15);
      float bias = bd[n];
#pragma unroll
      for (int e = 0; e < 8; ++e) {
        int m = mq * 16 + ((lane & 16) >> 1) + e;
        float v = acc[i][e] + bias;
        v = 0.5f * v * (1.0f + erff(v * 0.70710678118f));   // exact GELU
        xmid[(size_t)(mbase + m) * 128 + n] = v;
      }
    }
  }
}

// ---------------------------------------------------------------------------
// Kernel 2: fused SS2D pass, one block per image b.
// LDS map (dynamic, 110592 B):
//   [0      , 50176)  : phase1-2: Xh f16[64][128]  | phase5+: yb f32[49][256]
//   [50176  ,100352)  : xpc f16[49][256] + zb f16[49][256] | phase6+: yh f16[64][256]
//   [100352 ,110592)  : dtA f32[4][49][8], Bm/Cm f32[4][49], mu/rs f32[49], ps f32[49][4][2]
// ---------------------------------------------------------------------------
#define SSM_SMEM 110592

__global__ __launch_bounds__(256) void k_ssm(const float* __restrict__ in, float* __restrict__ out,
    const _Float16* __restrict__ inw_h, const _Float16* __restrict__ outw_h,
    const float* __restrict__ conv_w, const float* __restrict__ x_proj_w,
    const float* __restrict__ dt_proj_w, const float* __restrict__ dt_proj_b,
    const float* __restrict__ A_logs, const float* __restrict__ Ds,
    const float* __restrict__ ln_g, const float* __restrict__ ln_b) {
  extern __shared__ char smem[];
  _Float16* Xh  = (_Float16*)smem;                       // [64][128]
  float*    yb  = (float*)smem;                          // [49][256]
  _Float16* xpc = (_Float16*)(smem + 50176);             // [49][256] xp -> xc in place
  _Float16* zb  = (_Float16*)(smem + 50176 + 25088);     // [49][256]
  _Float16* yh  = (_Float16*)(smem + 50176);             // [64][256]
  float* dtA = (float*)(smem + 100352);                  // [4][49][8]
  float* Bm  = (float*)(smem + 100352 + 6272);           // [4][49]
  float* Cm  = (float*)(smem + 100352 + 7056);           // [4][49]
  float* mu  = (float*)(smem + 100352 + 7840);           // [49]
  float* rs  = (float*)(smem + 100352 + 8036);           // [49]
  float* ps  = (float*)(smem + 100352 + 8240);           // [49][4][2] LN partials

  const int tid = threadIdx.x;
  const int lane = tid & 31;
  const int wv = tid >> 5;
  const int b = blockIdx.x;
  const float* xin = in + (size_t)b * 49 * 128;

  // P1: load X (49x128) -> f16, zero-pad M to 64 (flat: idx = r*128+k)
  for (int i = tid * 4; i < 64 * 128; i += 1024) {
    float4 v;
    if (i < 49 * 128) v = *(const float4*)(xin + i);
    else              v = make_float4(0.f, 0.f, 0.f, 0.f);
    *(v4h*)(Xh + i) = cvt4(v.x, v.y, v.z, v.w);
  }
  __syncthreads();

  // P2: in-proj WMMA: XZ(64x512) = Xh(64x128) @ in_w^T; split into xp | z
  {
    int n0 = wv * 64;
    for (int m = 0; m < 4; ++m) {
      v8f acc[4] = {};
      for (int kk = 0; kk < 128; kk += 32) {
        v16h a = frag_ld(Xh + m * 16 * 128 + kk, 128, lane);
        v16h bf[4];
#pragma unroll
        for (int i = 0; i < 4; ++i)
          bf[i] = frag_ld_k(inw_h + (size_t)(n0 + i * 16 + (lane & 15)) * 128 + kk, lane);
#pragma unroll
        for (int i = 0; i < 4; ++i) acc[i] = WMMA_F16(a, bf[i], acc[i]);
      }
#pragma unroll
      for (int i = 0; i < 4; ++i) {
        int n = n0 + i * 16 + (lane & 15);
#pragma unroll
        for (int e = 0; e < 8; ++e) {
          int mr = m * 16 + ((lane & 16) >> 1) + e;
          if (mr < 49) {
            _Float16 v = (_Float16)acc[i][e];
            if (n < 256) xpc[mr * 256 + n] = v;
            else         zb[mr * 256 + (n - 256)] = v;
          }
        }
      }
    }
  }
  __syncthreads();

  // P3: depthwise 3x3 SAME conv + silu, in place (thread owns channel d)
  {
    int d = tid;
    float v[49];
#pragma unroll
    for (int l = 0; l < 49; ++l) v[l] = (float)xpc[l * 256 + d];
    float w9[9];
#pragma unroll
    for (int t = 0; t < 9; ++t) w9[t] = conv_w[t * 256 + d];
#pragma unroll
    for (int hi = 0; hi < 7; ++hi)
#pragma unroll
      for (int wi = 0; wi < 7; ++wi) {
        float s = 0.f;
#pragma unroll
        for (int kh = 0; kh < 3; ++kh)
#pragma unroll
          for (int kw = 0; kw < 3; ++kw) {
            int ih = hi + kh - 1, iw = wi + kw - 1;
            if (ih >= 0 && ih < 7 && iw >= 0 && iw < 7)
              s += w9[kh * 3 + kw] * v[ih * 7 + iw];
          }
        s = s / (1.f + expf(-s));                       // silu
        xpc[(hi * 7 + wi) * 256 + d] = (_Float16)s;     // own column: no race
      }
  }
  __syncthreads();

  // P4: x-proj: x_dbl[k][c][l] = sum_d x_proj_w[k][c][d] * xc[pos_k(l)][d]
  for (int j = tid; j < 4 * 10 * 49; j += 256) {
    int k = j / 490;
    int rem = j - k * 490;
    int c = rem / 49;
    int l = rem - c * 49;
    int pos = pos_dir(k, l);
    const float* wrow = x_proj_w + (k * 10 + c) * 256;
    const _Float16* xc = xpc + pos * 256;
    float s = 0.f;
#pragma unroll 8
    for (int d = 0; d < 256; ++d) s += wrow[d] * (float)xc[d];
    if (c < 8)       dtA[(k * 49 + l) * 8 + c] = s;
    else if (c == 8) Bm[k * 49 + l] = s;
    else             Cm[k * 49 + l] = s;
  }
  for (int i = tid; i < 49 * 256; i += 256) yb[i] = 0.f;   // Xh dead -> y accum
  __syncthreads();

  // P5: 4-direction selective scan (D_STATE=1), thread owns channel d
  {
    int d = tid;
    for (int k = 0; k < 4; ++k) {
      float wt[8];
#pragma unroll
      for (int r = 0; r < 8; ++r) wt[r] = dt_proj_w[(k * 256 + d) * 8 + r];
      float bias = dt_proj_b[k * 256 + d];
      float A  = -expf(A_logs[k * 256 + d]);
      float Dv = Ds[k * 256 + d];
      float h = 0.f;
      for (int l = 0; l < 49; ++l) {
        const float* drow = dtA + (k * 49 + l) * 8;
        float dt = bias;
#pragma unroll
        for (int r = 0; r < 8; ++r) dt += wt[r] * drow[r];
        dt = (dt > 20.f) ? dt : log1pf(expf(dt));        // softplus
        float da = expf(dt * A);
        int pos = pos_dir(k, l);
        float xv = (float)xpc[pos * 256 + d];
        h = da * h + dt * Bm[k * 49 + l] * xv;
        float yv = h * Cm[k * 49 + l] + Dv * xv;
        yb[pos * 256 + d] += yv;                          // bijection per dir: race-free
      }
    }
  }
  __syncthreads();

  // P6: LayerNorm stats per position (parallel partials: 4 threads x 64 ch)
  if (tid < 196) {
    int l = tid >> 2, qq = tid & 3;
    const float* row = yb + l * 256 + qq * 64;
    float s = 0.f, ss = 0.f;
#pragma unroll 8
    for (int d = 0; d < 64; ++d) { float v = row[d]; s += v; ss += v * v; }
    ps[tid * 2 + 0] = s;
    ps[tid * 2 + 1] = ss;
  }
  __syncthreads();
  if (tid < 49) {
    float s = 0.f, ss = 0.f;
#pragma unroll
    for (int qq = 0; qq < 4; ++qq) { s += ps[(tid * 4 + qq) * 2]; ss += ps[(tid * 4 + qq) * 2 + 1]; }
    float m = s * (1.f / 256.f);
    mu[tid] = m;
    rs[tid] = rsqrtf(ss * (1.f / 256.f) - m * m + 1e-5f);
  }
  __syncthreads();
  // gate: yh = LN(y) * silu(z), rows<49 fit inside xpc region (z untouched)
  for (int i = tid; i < 49 * 256; i += 256) {
    int l = i >> 8, d = i & 255;
    float yn = (yb[i] - mu[l]) * rs[l] * ln_g[d] + ln_b[d];
    float zv = (float)zb[i];
    yh[i] = (_Float16)(yn * (zv / (1.f + expf(-zv))));
  }
  __syncthreads();
  for (int i = tid; i < 15 * 256; i += 256) yh[49 * 256 + i] = (_Float16)0.f;  // pad (z dead)
  __syncthreads();

  // P7: out-proj WMMA: OUT(64x128) = yh(64x256) @ out_w^T
  {
    int mq = wv & 3, nq = wv >> 2;
    v8f acc[4] = {};
    for (int kk = 0; kk < 256; kk += 32) {
      v16h a = frag_ld(yh + mq * 16 * 256 + kk, 256, lane);
      v16h bf[4];
#pragma unroll
      for (int i = 0; i < 4; ++i)
        bf[i] = frag_ld_k(outw_h + (size_t)(nq * 64 + i * 16 + (lane & 15)) * 256 + kk, lane);
#pragma unroll
      for (int i = 0; i < 4; ++i) acc[i] = WMMA_F16(a, bf[i], acc[i]);
    }
    float* yout = out + (size_t)b * 49 * 128;
#pragma unroll
    for (int i = 0; i < 4; ++i) {
      int n = nq * 64 + i * 16 + (lane & 15);
#pragma unroll
      for (int e = 0; e < 8; ++e) {
        int mr = mq * 16 + ((lane & 16) >> 1) + e;
        if (mr < 49) yout[mr * 128 + n] = acc[i][e];
      }
    }
  }
}

// ---------------------------------------------------------------------------
// Kernel 3: t = (1+alpha)*xmid + (1-alpha)*y2 ; out = x + t @ Wu^T + bu
// M = 50176 (64/block), N = 1024 (128/block), K = 128
// (xmid/y2 ~51MB total: L2-resident, so the 8x N-split re-read stays off HBM)
// ---------------------------------------------------------------------------
__global__ __launch_bounds__(256) void k_upproj(const float* __restrict__ x,
                                                const float* __restrict__ xmid,
                                                const float* __restrict__ y2,
                                                const _Float16* __restrict__ wu_h,
                                                const float* __restrict__ bu,
                                                const float* __restrict__ gate1,
                                                float* __restrict__ out) {
  __shared__ _Float16 As[64 * 128];
  const int tid = threadIdx.x;
  const int lane = tid & 31;
  const int wv = tid >> 5;
  const int mq = wv & 3, nq = wv >> 2;
  const int mbase = blockIdx.x * 64;
  const int nbase = blockIdx.y * 128;

  float alpha = 1.0f / (1.0f + expf(-gate1[0]));
  float ca = 1.0f + alpha, cb = 1.0f - alpha;

  const float* xm = xmid + (size_t)mbase * 128;
  const float* yy = y2 + (size_t)mbase * 128;
  for (int i = tid * 4; i < 64 * 128; i += 1024) {
    float4 a4 = *(const float4*)(xm + i);
    float4 b4 = *(const float4*)(yy + i);
    *(v4h*)(As + i) = cvt4(ca * a4.x + cb * b4.x, ca * a4.y + cb * b4.y,
                           ca * a4.z + cb * b4.z, ca * a4.w + cb * b4.w);
  }
  __syncthreads();

  v8f acc[4] = {};
  for (int kk = 0; kk < 128; kk += 32) {
    v16h a = frag_ld(As + mq * 16 * 128 + kk, 128, lane);
    v16h bf[4];
#pragma unroll
    for (int i = 0; i < 4; ++i)
      bf[i] = frag_ld_k(wu_h + (size_t)(nbase + nq * 64 + i * 16 + (lane & 15)) * 128 + kk, lane);
#pragma unroll
    for (int i = 0; i < 4; ++i) acc[i] = WMMA_F16(a, bf[i], acc[i]);
  }
#pragma unroll
  for (int i = 0; i < 4; ++i) {
    int n = nbase + nq * 64 + i * 16 + (lane & 15);
    float bias = bu[n];
#pragma unroll
    for (int e = 0; e < 8; ++e) {
      int m = mq * 16 + ((lane & 16) >> 1) + e;
      int gr = mbase + m;
      int b = gr / 49, s = gr - b * 49;
      size_t off = ((size_t)s * 1024 + b) * 1024 + n;    // (S,B,HID)
      out[off] = x[off] + acc[i][e] + bias;
    }
  }
}

// ---------------------------------------------------------------------------
// Workspace layout (bytes)
// ---------------------------------------------------------------------------
#define WS_WD_H   0u
#define WS_WU_H   262144u
#define WS_INW_H  524288u
#define WS_OUTW_H 655360u
#define WS_XMID   720896u
#define WS_Y1     (720896u + 25690112u)
#define WS_Y2     (720896u + 2u * 25690112u)

extern "C" void kernel_launch(void* const* d_in, const int* in_sizes, int n_in,
                              void* d_out, int out_size, void* d_ws, size_t ws_size,
                              hipStream_t stream) {
  const float* x         = (const float*)d_in[0];
  const float* Wd        = (const float*)d_in[1];
  const float* bd        = (const float*)d_in[2];
  const float* Wu        = (const float*)d_in[3];
  const float* bu        = (const float*)d_in[4];
  const float* gate1     = (const float*)d_in[5];
  const float* in_w      = (const float*)d_in[6];
  const float* conv_w    = (const float*)d_in[7];
  const float* x_proj_w  = (const float*)d_in[8];
  const float* dt_proj_w = (const float*)d_in[9];
  const float* dt_proj_b = (const float*)d_in[10];
  const float* A_logs    = (const float*)d_in[11];
  const float* Ds        = (const float*)d_in[12];
  const float* ln_g      = (const float*)d_in[13];
  const float* ln_b      = (const float*)d_in[14];
  const float* out_w     = (const float*)d_in[15];

  char* ws = (char*)d_ws;
  _Float16* wd_h   = (_Float16*)(ws + WS_WD_H);
  _Float16* wu_h   = (_Float16*)(ws + WS_WU_H);
  _Float16* inw_h  = (_Float16*)(ws + WS_INW_H);
  _Float16* outw_h = (_Float16*)(ws + WS_OUTW_H);
  float* xmid = (float*)(ws + WS_XMID);
  float* y1   = (float*)(ws + WS_Y1);
  float* y2   = (float*)(ws + WS_Y2);

  k_wconv<<<512, 256, 0, stream>>>(Wd, Wu, in_w, out_w, wd_h, wu_h, inw_h, outw_h);
  k_downproj<<<784, 256, 0, stream>>>(x, wd_h, bd, xmid);
  k_ssm<<<1024, 256, SSM_SMEM, stream>>>(xmid, y1, inw_h, outw_h, conv_w, x_proj_w,
                                         dt_proj_w, dt_proj_b, A_logs, Ds, ln_g, ln_b);
  k_ssm<<<1024, 256, SSM_SMEM, stream>>>(y1, y2, inw_h, outw_h, conv_w, x_proj_w,
                                         dt_proj_w, dt_proj_b, A_logs, Ds, ln_g, ln_b);
  k_upproj<<<dim3(784, 8), 256, 0, stream>>>(x, xmid, y2, wu_h, bu, gate1, (float*)d_out);
}
